// MultiGroupQuantizer_11098195493291
// MI455X (gfx1250) — compile-verified, matrix operand order checked by reading, not run on hardware
//
#include <hip/hip_runtime.h>

// ---------------------------------------------------------------------------
// MultiGroupQuantizer for MI455X (gfx1250, wave32, WMMA)
// dist(m,k) = ||x_m||^2 - 2 x_m.c_k + ||c_k||^2  -> argmin_k ( 0.5||c_k||^2 - x_m.c_k )
// GEMM x.c^T done with v_wmma_f32_16x16x32_bf16 using a bf16 hi/lo split:
//   x ~ xh + xl, c ~ ch + cl,  x.c ~ xh.ch + xh.cl + xl.ch   (3 WMMAs / k-chunk)
// B stream is software-pipelined 2 chunks ahead (double buffer).
// ---------------------------------------------------------------------------

typedef __attribute__((ext_vector_type(16))) __bf16 v16bf;
typedef __attribute__((ext_vector_type(8)))  __bf16 v8bf;
typedef __attribute__((ext_vector_type(8)))  float  v8f;

union V16U { v16bf v; v8bf h[2]; };

#define N_BATCH   64
#define WIDTH     512
#define T_LEN     1024
#define D_DIM     256          // per-group channel dim
#define M_TOK     (N_BATCH * T_LEN)   // 65536 tokens per group
#define K0        512
#define K1        1024

#define MTILE     128          // rows per workgroup (8 waves x 16)
#define XROW      264          // padded LDS row stride in bf16 elems (16B aligned)
#define SMEM_BYTES (2 * MTILE * XROW * 2)  // hi + lo, 2B each = 135168

// d_out layout (floats)
#define OUT_X      0
#define OUT_COMMIT ((size_t)N_BATCH * WIDTH * T_LEN)          // 33554432
#define OUT_PERP   (OUT_COMMIT + 1)
#define OUT_IDX0   (OUT_PERP + 2)
#define OUT_IDX1   (OUT_IDX0 + M_TOK)

// workspace layout (bytes)
#define WS_CB0_HI  0
#define WS_CB0_LO  (WS_CB0_HI + K0 * D_DIM * 2)
#define WS_CB1_HI  (WS_CB0_LO + K0 * D_DIM * 2)
#define WS_CB1_LO  (WS_CB1_HI + K1 * D_DIM * 2)
#define WS_CNORM0  (WS_CB1_LO + K1 * D_DIM * 2)
#define WS_CNORM1  (WS_CNORM0 + K0 * 4)
#define WS_IDX0    (WS_CNORM1 + K1 * 4)
#define WS_IDX1    (WS_IDX0 + M_TOK * 4)
#define WS_COUNTS0 (WS_IDX1 + M_TOK * 4)
#define WS_COUNTS1 (WS_COUNTS0 + K0 * 4)
#define WS_COMMIT  (WS_COUNTS1 + K1 * 4)
#define WS_ZERO_BEGIN WS_COUNTS0
#define WS_ZERO_BYTES (K0 * 4 + K1 * 4 + 256)

static __device__ __forceinline__ v8f wmma_bf16(v16bf a, v16bf b, v8f c) {
    return __builtin_amdgcn_wmma_f32_16x16x32_bf16(false, a, false, b,
                                                   (short)0, c, false, false);
}

static __device__ __forceinline__ float block_reduce_sum256(float v) {
    __shared__ float sh[256];
    int tid = threadIdx.x;
    sh[tid] = v;
    __syncthreads();
    for (int s = 128; s > 0; s >>= 1) {
        if (tid < s) sh[tid] += sh[tid + s];
        __syncthreads();
    }
    return sh[0];
}

// ---------------------------------------------------------------------------
// Kernel 1: split codebook fp32 -> bf16 hi/lo, precompute 0.5*||c||^2
// grid = K blocks, 256 threads (one per dim)
// ---------------------------------------------------------------------------
__global__ void prep_cb_kernel(const float* __restrict__ cb,
                               __bf16* __restrict__ hi,
                               __bf16* __restrict__ lo,
                               float* __restrict__ cnorm) {
    int k = blockIdx.x;
    int j = threadIdx.x;                 // 0..255
    float v = cb[(size_t)k * D_DIM + j];
    __bf16 h = (__bf16)v;
    float  hv = (float)h;
    __bf16 l = (__bf16)(v - hv);
    hi[(size_t)k * D_DIM + j] = h;
    lo[(size_t)k * D_DIM + j] = l;
    float s = block_reduce_sum256(v * v);
    if (j == 0) cnorm[k] = 0.5f * s;
}

// ---------------------------------------------------------------------------
// Kernel 2: WMMA distance GEMM + running argmin
// grid = M_TOK/MTILE = 512 blocks, 256 threads (8 waves x 16 rows)
// ---------------------------------------------------------------------------
__global__ void __launch_bounds__(256, 1)
quantize_kernel(const float* __restrict__ x,
                const __bf16* __restrict__ cbhi,
                const __bf16* __restrict__ cblo,
                const float* __restrict__ cnorm,
                int K, int chanStart,
                int* __restrict__ idx_int,
                float* __restrict__ idx_f,
                unsigned int* __restrict__ counts) {
    extern __shared__ __bf16 smem[];
    __bf16* xhi = smem;
    __bf16* xlo = smem + MTILE * XROW;

    const int tid  = threadIdx.x;
    const int wave = tid >> 5;
    const int lane = tid & 31;

    const int m0 = blockIdx.x * MTILE;     // 1024 % 128 == 0 -> tile stays in one n
    const int n  = m0 >> 10;
    const int t0 = m0 & 1023;

    // ---- stage x tile fp32 -> bf16 hi/lo in LDS (coalesced along t) ----
    const size_t xbase = (size_t)n * (WIDTH * T_LEN) + (size_t)chanStart * T_LEN + t0;
    #pragma unroll 4
    for (int e = tid; e < MTILE * D_DIM; e += 256) {
        int j  = e >> 7;                   // 0..255 channel
        int tl = e & 127;                  // 0..127 token-in-tile
        float v = x[xbase + (size_t)j * T_LEN + tl];
        __bf16 h = (__bf16)v;
        float  hv = (float)h;
        xhi[tl * XROW + j] = h;
        xlo[tl * XROW + j] = (__bf16)(v - hv);
    }
    __syncthreads();

    // ---- load this wave's A fragments (16 rows x 256 dims, hi & lo) ----
    const int rowA = wave * 16 + (lane & 15);
    const int ksel = (lane >> 4) * 8;      // K sub-chunk select per ISA A layout
    V16U ahi[8], alo[8];
    #pragma unroll
    for (int c = 0; c < 8; ++c) {
        const __bf16* ph = &xhi[rowA * XROW + c * 32 + ksel];
        const __bf16* pl = &xlo[rowA * XROW + c * 32 + ksel];
        ahi[c].h[0] = *(const v8bf*)(ph);
        ahi[c].h[1] = *(const v8bf*)(ph + 16);
        alo[c].h[0] = *(const v8bf*)(pl);
        alo[c].h[1] = *(const v8bf*)(pl + 16);
    }

    // ---- sweep codebook in 16-code tiles; per-lane running argmin ----
    float bestv[8];
    int   besti[8];
    #pragma unroll
    for (int r = 0; r < 8; ++r) { bestv[r] = 3.4e38f; besti[r] = 0; }

    const int laneCol = lane & 15;

// load B hi/lo fragments for k-chunk (c) into a double-buffer slot
#define LOADB(H, L, c) do {                                         \
        (H).h[0] = *(const v8bf*)(cbhi + boff + (c) * 32);          \
        (H).h[1] = *(const v8bf*)(cbhi + boff + (c) * 32 + 16);     \
        (L).h[0] = *(const v8bf*)(cblo + boff + (c) * 32);          \
        (L).h[1] = *(const v8bf*)(cblo + boff + (c) * 32 + 16);     \
    } while (0)

// three-term hi/lo WMMA accumulate for k-chunk (c)
#define WMMA3(c, H, L) do {                                         \
        acc = wmma_bf16(ahi[c].v, (H).v, acc);                      \
        acc = wmma_bf16(ahi[c].v, (L).v, acc);                      \
        acc = wmma_bf16(alo[c].v, (H).v, acc);                      \
    } while (0)

    for (int tb = 0; tb < K; tb += 16) {
        const int myCode = tb + laneCol;
        // 32-bit element offset from uniform base -> SGPR base + VGPR offset
        const int boff = myCode * D_DIM + ksel;
        const float cn = cnorm[myCode];

        v8f acc = {};
        V16U BhA, BlA, BhB, BlB;
        // prologue: two chunks in flight before first consume
        LOADB(BhA, BlA, 0);
        LOADB(BhB, BlB, 1);
        WMMA3(0, BhA, BlA);  LOADB(BhA, BlA, 2);
        WMMA3(1, BhB, BlB);  LOADB(BhB, BlB, 3);
        WMMA3(2, BhA, BlA);  LOADB(BhA, BlA, 4);
        WMMA3(3, BhB, BlB);  LOADB(BhB, BlB, 5);
        WMMA3(4, BhA, BlA);  LOADB(BhA, BlA, 6);
        WMMA3(5, BhB, BlB);  LOADB(BhB, BlB, 7);
        WMMA3(6, BhA, BlA);
        WMMA3(7, BhB, BlB);

        #pragma unroll
        for (int r = 0; r < 8; ++r) {
            float d = cn - acc[r];                  // 0.5||c||^2 - x.c
            if (d < bestv[r]) { bestv[r] = d; besti[r] = myCode; }
        }
    }
#undef LOADB
#undef WMMA3

    // ---- cross-lane argmin over the 16 candidate columns in each half ----
    #pragma unroll
    for (int off = 8; off >= 1; off >>= 1) {
        #pragma unroll
        for (int r = 0; r < 8; ++r) {
            float ov = __shfl_xor(bestv[r], off, 32);
            int   oi = __shfl_xor(besti[r], off, 32);
            if (ov < bestv[r] || (ov == bestv[r] && oi < besti[r])) {
                bestv[r] = ov; besti[r] = oi;       // first-occurrence tie-break
            }
        }
    }

    if ((lane & 15) == 0) {
        const int half = lane >> 4;                 // C rows: half*8 + r
        #pragma unroll
        for (int r = 0; r < 8; ++r) {
            int m  = m0 + wave * 16 + half * 8 + r;
            int bi = besti[r];
            idx_int[m] = bi;
            idx_f[m]   = (float)bi;
            atomicAdd(&counts[bi], 1u);
        }
    }
}

// ---------------------------------------------------------------------------
// Kernel 3: gather q = cb[idx] into [N, C, T] output + commitment sum
// grid = M_TOK/64 = 1024 blocks, 256 threads
// ---------------------------------------------------------------------------
__global__ void epilogue_kernel(const float* __restrict__ x,
                                const float* __restrict__ cb,
                                const int* __restrict__ idx_int,
                                float* __restrict__ out_x,
                                float* __restrict__ commit_accum,
                                int chanStart) {
    const int tid = threadIdx.x;
    const int m0  = blockIdx.x * 64;
    const int n   = m0 >> 10;
    const int t0  = m0 & 1023;
    const int tl  = tid & 63;
    const int jq  = tid >> 6;              // 0..3

    const int idx = idx_int[m0 + tl];
    const float* crow = cb + (size_t)idx * D_DIM;
    const size_t base = (size_t)n * (WIDTH * T_LEN) + (size_t)t0 + tl;

    float acc = 0.0f;
    #pragma unroll 8
    for (int jj = 0; jj < 64; ++jj) {
        int j = jq * 64 + jj;
        size_t o = base + (size_t)(chanStart + j) * T_LEN;
        float cv = crow[j];
        float df = x[o] - cv;
        acc += df * df;
        out_x[o] = cv;                     // straight-through forward value
    }
    float s = block_reduce_sum256(acc);
    if (tid == 0) atomicAdd(commit_accum, s);
}

// ---------------------------------------------------------------------------
// Kernel 4: perplexity per group + commitment scalar.  grid = 2 blocks.
// ---------------------------------------------------------------------------
__global__ void finalize_kernel(const unsigned int* __restrict__ counts0,
                                const unsigned int* __restrict__ counts1,
                                const float* __restrict__ commit_sums,
                                float* __restrict__ out_commit,
                                float* __restrict__ out_perp) {
    const int g   = blockIdx.x;
    const int tid = threadIdx.x;
    const unsigned int* counts = g ? counts1 : counts0;
    const int K = g ? K1 : K0;

    float h = 0.0f;
    for (int k = tid; k < K; k += 256) {
        float p = (float)counts[k] * (1.0f / (float)M_TOK);
        h += p * logf(p + 1e-7f);
    }
    float H = block_reduce_sum256(h);
    if (tid == 0) {
        out_perp[g] = expf(-H);
        if (g == 0)
            out_commit[0] = (commit_sums[0] + commit_sums[1]) *
                            (1.0f / ((float)M_TOK * (float)D_DIM));
    }
}

// ---------------------------------------------------------------------------
extern "C" void kernel_launch(void* const* d_in, const int* in_sizes, int n_in,
                              void* d_out, int out_size, void* d_ws, size_t ws_size,
                              hipStream_t stream) {
    (void)in_sizes; (void)n_in; (void)out_size; (void)ws_size;
    const float* x   = (const float*)d_in[0];
    const float* cb0 = (const float*)d_in[1];
    const float* cb1 = (const float*)d_in[2];
    float* out = (float*)d_out;
    char*  ws  = (char*)d_ws;

    __bf16* cb0hi = (__bf16*)(ws + WS_CB0_HI);
    __bf16* cb0lo = (__bf16*)(ws + WS_CB0_LO);
    __bf16* cb1hi = (__bf16*)(ws + WS_CB1_HI);
    __bf16* cb1lo = (__bf16*)(ws + WS_CB1_LO);
    float*  cn0   = (float*)(ws + WS_CNORM0);
    float*  cn1   = (float*)(ws + WS_CNORM1);
    int*    idx0  = (int*)(ws + WS_IDX0);
    int*    idx1  = (int*)(ws + WS_IDX1);
    unsigned int* cnt0 = (unsigned int*)(ws + WS_COUNTS0);
    unsigned int* cnt1 = (unsigned int*)(ws + WS_COUNTS1);
    float*  commit = (float*)(ws + WS_COMMIT);

    // zero counts + commitment accumulators (graph-capture safe)
    hipMemsetAsync(ws + WS_ZERO_BEGIN, 0, WS_ZERO_BYTES, stream);

    prep_cb_kernel<<<K0, 256, 0, stream>>>(cb0, cb0hi, cb0lo, cn0);
    prep_cb_kernel<<<K1, 256, 0, stream>>>(cb1, cb1hi, cb1lo, cn1);

    quantize_kernel<<<M_TOK / MTILE, 256, SMEM_BYTES, stream>>>(
        x, cb0hi, cb0lo, cn0, K0, 0,   idx0, out + OUT_IDX0, cnt0);
    quantize_kernel<<<M_TOK / MTILE, 256, SMEM_BYTES, stream>>>(
        x, cb1hi, cb1lo, cn1, K1, 256, idx1, out + OUT_IDX1, cnt1);

    epilogue_kernel<<<M_TOK / 64, 256, 0, stream>>>(x, cb0, idx0, out, commit + 0, 0);
    epilogue_kernel<<<M_TOK / 64, 256, 0, stream>>>(x, cb1, idx1, out, commit + 1, 256);

    finalize_kernel<<<2, 256, 0, stream>>>(cnt0, cnt1, commit,
                                           out + OUT_COMMIT, out + OUT_PERP);
}
